// TauTwoStepSimple_31808527794729
// MI455X (gfx1250) — matrix operand order repaired and checked by reading, not compiled, not run on hardware
//
#include <hip/hip_runtime.h>
#include <hip/hip_bf16.h>
#include <stdint.h>

// ---------------------------------------------------------------------------
// TauTwoStepSimple inference for MI455X (gfx1250, wave32, WMMA).
// All GEMMs: v_wmma_f32_16x16x32_f16. One wave computes a 32x(16*NT) output
// tile (2 A-fragments x NT B-fragments -> 2*NT WMMAs per K-step). NT=4 for
// 64-wide column blocks (heavy layers), NT=2 fallback for 32-wide outputs.
// Compute-bound (~400 GFLOP); edge tensors materialized per event-chunk so
// the GEMM inner loop is pure b128 loads + WMMA (no divergence).
// ---------------------------------------------------------------------------

typedef _Float16 half_t;
typedef __attribute__((ext_vector_type(16))) _Float16 v16h;
typedef __attribute__((ext_vector_type(8)))  _Float16 v8h;
typedef __attribute__((ext_vector_type(8)))  float    v8f;

#define NPART 20
#define NFEAT 14
#define NEVT  1024
#define EPSV  1e-5f

// ------------------------------- kernels -----------------------------------

__global__ void bn_input_kernel(const float* __restrict__ x,
                                const float* __restrict__ beta, const float* __restrict__ gamma,
                                const float* __restrict__ mean, const float* __restrict__ var,
                                float* __restrict__ y, int total) {
  int i = blockIdx.x * blockDim.x + threadIdx.x;
  if (i >= total) return;
  int c = i % NFEAT;
  float inv = rsqrtf(var[c] + EPSV);
  y[i] = (x[i] - mean[c]) * inv * gamma[c] + beta[c];
}

// one wave per event: brute-force 20x20 d2 + k-smallest selection per row
__global__ void knn_kernel(const float* __restrict__ pts, int D, int kE,
                           int* __restrict__ nbr) {
  __shared__ float sp[NPART * 256];
  __shared__ float d2[NPART * NPART];
  int b = blockIdx.x, lane = threadIdx.x;
  const float* P = pts + (size_t)b * NPART * D;
  for (int t = lane; t < NPART * D; t += 32) sp[t] = P[t];
  __syncthreads();
  for (int p = lane; p < NPART * NPART; p += 32) {
    int i = p / NPART, j = p % NPART;
    float s = 0.f;
    for (int c = 0; c < D; ++c) { float d = sp[i * D + c] - sp[j * D + c]; s += d * d; }
    d2[p] = (i == j) ? 1e30f : s;
  }
  __syncthreads();
  if (lane < NPART) {
    float row[NPART];
    for (int j = 0; j < NPART; ++j) row[j] = d2[lane * NPART + j];
    for (int t = 0; t < kE; ++t) {
      int bj = 0; float bv = 3.4e38f;
      for (int j = 0; j < NPART; ++j) if (row[j] < bv) { bv = row[j]; bj = j; }
      nbr[((size_t)b * NPART + lane) * kE + t] = bj;
      row[bj] = 3.4e38f;
    }
  }
}

// W f32 [K,Nv] row-major -> Wt f16 [Np,Kp] (transposed, zero padded)
__global__ void convert_w_kernel(const float* __restrict__ W, half_t* __restrict__ Wt,
                                 int K, int Nv, int Kp, int Np, int total) {
  int i = blockIdx.x * blockDim.x + threadIdx.x;
  if (i >= total) return;
  int n = i / Kp, k = i % Kp;
  Wt[i] = (n < Nv && k < K) ? (half_t)W[(size_t)k * Nv + n] : (half_t)0.f;
}

// f32 [rows,L] -> f16 [rows,Lp] zero-padded
__global__ void pack_half_kernel(const float* __restrict__ X, int L,
                                 half_t* __restrict__ Y, int Lp, int total) {
  int i = blockIdx.x * blockDim.x + threadIdx.x;
  if (i >= total) return;
  int r = i / Lp, c = i % Lp;
  Y[i] = (c < L) ? (half_t)X[(size_t)r * L + c] : (half_t)0.f;
}

// Materialize edge features H[r, c] = [x_i | x_j - x_i] in f16, zero-padded to Kp.
// r = node*kE + kk (chunk-local), nbr holds local neighbor index 0..19.
__global__ void edge_build_kernel(const float* __restrict__ fts, const int* __restrict__ nbr,
                                  int Cin, int kE, half_t* __restrict__ H, int Kp, int total) {
  int i = blockIdx.x * blockDim.x + threadIdx.x;
  if (i >= total) return;
  int r = i / Kp, c = i % Kp;
  int node = r / kE;
  float v = 0.f;
  if (c < Cin) {
    v = fts[(size_t)node * Cin + c];
  } else if (c < 2 * Cin) {
    int jl = nbr[r];
    int jn = (node / NPART) * NPART + jl;
    int cc = c - Cin;
    v = fts[(size_t)jn * Cin + cc] - fts[(size_t)node * Cin + cc];
  }
  H[i] = (half_t)v;
}

// ---------------------------------------------------------------------------
// WMMA GEMM: D[M,Np] = A[M,Kp] x Wt^T   (A f16 [M,lda], Wt f16 [Np,Kp])
// One wave per 32x(16*NT) output tile: 2 A-frags x NT B-frags, 2*NT WMMAs/step.
// epi: 0 BN+ReLU (E0=beta,E1=gamma,E2=mean,E3=var)  1 BN
//      2 bias+ELU (E0=bias)                          3 bias
// ---------------------------------------------------------------------------
template <int NT>
__global__ void wmma_gemm_kernel(
    const half_t* __restrict__ A, int lda,
    const half_t* __restrict__ Wt, int ldw, int Kp, int Np,
    float* __restrict__ C32, int ldc32,
    half_t* __restrict__ C16, int ldc16,
    int epi, const float* __restrict__ E0, const float* __restrict__ E1,
    const float* __restrict__ E2, const float* __restrict__ E3,
    int Nvalid) {
  int lane = threadIdx.x;
  int tm = blockIdx.x, tn = blockIdx.y;
  int l15 = lane & 15;
  int r0 = tm * 32 + l15;            // A fragment rows (both half-waves M=0..15)
  int r1 = r0 + 16;
  int ksel = (lane < 16) ? 0 : 16;   // A: half-wave 1 holds K+16..31
  int bsel = (lane < 16) ? 0 : 16;   // B: half-wave 1 holds K+16..31

  const half_t* a0p = A + (size_t)r0 * lda + ksel;
  const half_t* a1p = A + (size_t)r1 * lda + ksel;
  const half_t* bp[NT];
  int ncol[NT];
#pragma unroll
  for (int j = 0; j < NT; ++j) {
    ncol[j] = tn * (16 * NT) + j * 16 + l15;
    bp[j] = Wt + (size_t)ncol[j] * ldw + bsel;
  }

  v8f acc0[NT], acc1[NT];
#pragma unroll
  for (int j = 0; j < NT; ++j) { acc0[j] = (v8f){}; acc1[j] = (v8f){}; }

  for (int k0 = 0; k0 < Kp; k0 += 32) {
    if (k0 + 32 < Kp) {  // prefetch next K-step (global_prefetch_b8)
      __builtin_prefetch(a0p + k0 + 32, 0, 1);
      __builtin_prefetch(a1p + k0 + 32, 0, 1);
#pragma unroll
      for (int j = 0; j < NT; ++j) __builtin_prefetch(bp[j] + k0 + 32, 0, 1);
    }
    v16h a0, a1;
    {
      v8h lo = *(const v8h*)(a0p + k0);
      v8h hi = *(const v8h*)(a0p + k0 + 8);
#pragma unroll
      for (int e = 0; e < 8; ++e) { a0[e] = lo[e]; a0[8 + e] = hi[e]; }
    }
    {
      v8h lo = *(const v8h*)(a1p + k0);
      v8h hi = *(const v8h*)(a1p + k0 + 8);
#pragma unroll
      for (int e = 0; e < 8; ++e) { a1[e] = lo[e]; a1[8 + e] = hi[e]; }
    }
#pragma unroll
    for (int j = 0; j < NT; ++j) {
      v16h b = *(const v16h*)(bp[j] + k0);
      acc0[j] = __builtin_amdgcn_wmma_f32_16x16x32_f16(false, a0, false, b, (short)0, acc0[j], false, false);
      acc1[j] = __builtin_amdgcn_wmma_f32_16x16x32_f16(false, a1, false, b, (short)0, acc1[j], false, false);
    }
  }

  int act = (epi == 0) ? 1 : (epi == 2) ? 2 : 0;
  int rb = ((lane < 16) ? 0 : 8);  // C/D layout: VGPR r -> rows r / r+8
#pragma unroll
  for (int j = 0; j < NT; ++j) {
    int n = ncol[j];
    float sc = 1.f, sh = 0.f;
    if (epi == 0 || epi == 1) {
      if (n < Nvalid) {
        float inv = rsqrtf(E3[n] + EPSV);
        sc = E1[n] * inv;
        sh = E0[n] - E2[n] * sc;
      } else { sc = 0.f; }
    } else {
      if (n < Nvalid) sh = E0[n]; else sc = 0.f;
    }
#pragma unroll
    for (int half = 0; half < 2; ++half) {
      const v8f& acc = half ? acc1[j] : acc0[j];
      int rowBase = tm * 32 + half * 16 + rb;
#pragma unroll
      for (int r = 0; r < 8; ++r) {
        float v = acc[r] * sc + sh;
        if (act == 1)      v = fmaxf(v, 0.f);
        else if (act == 2) v = (v > 0.f) ? v : (__expf(v) - 1.f);
        int row = rowBase + r;
        if (C32) C32[(size_t)row * ldc32 + n] = v;
        if (C16) C16[(size_t)row * ldc16 + n] = (half_t)v;
      }
    }
  }
}

// relu(max_k(mlp) + skip) -> fts_next (stride c2)
__global__ void aggr_kernel(const half_t* __restrict__ h3, int ldh, int kE, int c2,
                            const float* __restrict__ skip, int lds_,
                            float* __restrict__ ftsN, int node0, int total) {
  int i = blockIdx.x * blockDim.x + threadIdx.x;
  if (i >= total) return;
  int nl = i / c2, c = i % c2;
  float m = -3.4e38f;
  for (int kk = 0; kk < kE; ++kk)
    m = fmaxf(m, (float)h3[((size_t)nl * kE + kk) * ldh + c]);
  int ng = node0 + nl;
  float v = m + skip[(size_t)ng * lds_ + c];
  ftsN[(size_t)ng * c2 + c] = fmaxf(v, 0.f);
}

// nxt[b,680] = concat(raw feats[b,20,14], fts[b,20,20]) along last axis
__global__ void build_nxt_kernel(const float* __restrict__ feats,
                                 const float* __restrict__ fts,
                                 float* __restrict__ nxt, int total) {
  int i = blockIdx.x * blockDim.x + threadIdx.x;
  if (i >= total) return;
  int b = i / 680, t = i % 680;
  int p = t / 34,  c = t % 34;
  float v;
  if (c < NFEAT) v = feats[((size_t)b * NPART + p) * NFEAT + c];
  else           v = fts[((size_t)b * NPART + p) * NPART + (c - NFEAT)];
  nxt[i] = v;
}

// one wave per row; ldo >= L, pads with zeros. out_half: f16 out, else f32.
__global__ void layernorm_kernel(const float* __restrict__ X, int L, int ldx,
                                 const float* __restrict__ g, const float* __restrict__ b,
                                 void* __restrict__ out, int ldo, int out_half, int elu) {
  int row = blockIdx.x, lane = threadIdx.x;
  const float* xr = X + (size_t)row * ldx;
  float s = 0.f, s2 = 0.f;
  for (int c = lane; c < L; c += 32) { float v = xr[c]; s += v; s2 += v * v; }
  for (int off = 16; off > 0; off >>= 1) {
    s  += __shfl_xor(s,  off, 32);
    s2 += __shfl_xor(s2, off, 32);
  }
  float mean = s / (float)L;
  float var  = s2 / (float)L - mean * mean;
  float inv  = rsqrtf(var + EPSV);
  for (int c = lane; c < ldo; c += 32) {
    float v = 0.f;
    if (c < L) {
      v = (xr[c] - mean) * inv * g[c] + b[c];
      if (elu) v = (v > 0.f) ? v : (__expf(v) - 1.f);
    }
    if (out_half) ((half_t*)out)[(size_t)row * ldo + c] = (half_t)v;
    else          ((float*)out)[(size_t)row * ldo + c]  = v;
  }
}

__global__ void finalize_kernel(const float* __restrict__ fts,   // [B*20, 20]
                                const float* __restrict__ istau, // [B,32]
                                const float* __restrict__ p4,    // [B,32]
                                const float* __restrict__ charge,// [B,32]
                                const float* __restrict__ dmode, // [B,32]
                                float* __restrict__ out) {
  int b = blockIdx.x, lane = threadIdx.x;
  if (lane < NPART) {
    float m = -3.4e38f;
    for (int i = 0; i < NPART; ++i)
      m = fmaxf(m, fts[((size_t)b * NPART + i) * NPART + lane]);
    out[(size_t)b * NPART + lane] = 1.f / (1.f + __expf(-m));
  } else if (lane == 20) {
    float v = istau[(size_t)b * 32];
    out[20480 + b] = 1.f / (1.f + __expf(-v));
  } else if (lane >= 21 && lane < 25) {
    int c = lane - 21;
    out[21504 + (size_t)b * 4 + c] = 200.f * p4[(size_t)b * 32 + c];
  } else if (lane == 25) {
    float a0 = charge[(size_t)b * 32], a1 = charge[(size_t)b * 32 + 1];
    float mx = fmaxf(a0, a1);
    float e0 = __expf(a0 - mx), e1 = __expf(a1 - mx);
    float s = e0 + e1;
    out[25600 + (size_t)b * 2]     = e0 / s;
    out[25600 + (size_t)b * 2 + 1] = e1 / s;
  } else if (lane == 26) {
    float e[6]; float mx = -3.4e38f;
    for (int c = 0; c < 6; ++c) mx = fmaxf(mx, dmode[(size_t)b * 32 + c]);
    float s = 0.f;
    for (int c = 0; c < 6; ++c) { e[c] = __expf(dmode[(size_t)b * 32 + c] - mx); s += e[c]; }
    for (int c = 0; c < 6; ++c) out[27648 + (size_t)b * 6 + c] = e[c] / s;
  }
}

// ------------------------------- host --------------------------------------

static inline int pad32i(int x) { return (x + 31) & ~31; }

// Parameter leaf indices assume JAX pytree flattening (dicts by sorted key):
//   0 gnnfeats, 1 gnnpos
//   2..5    bn_in: beta,gamma,mean,var
//   6..27   charge FFN: lin W0,b0..W4,b4 (6..15); ln g0,b0..g5,b5 (16..27)
//   28+20*c convs[c]: mlp0 W,(beta,gamma,mean,var), mlp1 W,(..), mlp2 W,(..),
//                     skip_bn beta,gamma,mean,var, skip_w
//   128..149 dmode FFN   150..171 enc FFN   172..193 istau FFN   194..215 p4 FFN
extern "C" void kernel_launch(void* const* d_in, const int* in_sizes, int n_in,
                              void* d_out, int out_size, void* d_ws, size_t ws_size,
                              hipStream_t stream) {
  (void)in_sizes; (void)out_size;
  if (n_in < 216) return;
  const float* gnnfeats = (const float*)d_in[0];
  const float* gnnpos   = (const float*)d_in[1];
  auto P = [&](int i) -> const float* { return (const float*)d_in[i]; };

  uintptr_t base = (uintptr_t)d_ws;
  uintptr_t cur  = base;
  auto alloc = [&](size_t bytes) -> void* {
    uintptr_t p = (cur + 255) & ~(uintptr_t)255;
    cur = p + bytes;
    return (void*)p;
  };

  const int NODES = NEVT * NPART; // 20480
  float*  fts_a  = (float*)alloc((size_t)NODES * 256 * 4);
  float*  fts_b  = (float*)alloc((size_t)NODES * 256 * 4);
  float*  skip   = (float*)alloc((size_t)NODES * 256 * 4);
  half_t* act16  = (half_t*)alloc((size_t)NODES * 256 * 2);
  int*    nbr    = (int*)alloc((size_t)NODES * 19 * 4);
  float*  nxt    = (float*)alloc((size_t)NEVT * 680 * 4);
  half_t* t16    = (half_t*)alloc((size_t)NEVT * 704 * 2);
  float*  x0     = (float*)alloc((size_t)NEVT * 128 * 4);
  float*  x1     = (float*)alloc((size_t)NEVT * 128 * 4);
  float*  x2     = (float*)alloc((size_t)NEVT * 128 * 4);
  float*  enc_o  = (float*)alloc((size_t)NEVT * 128 * 4);
  float*  hb_is  = (float*)alloc((size_t)NEVT * 32 * 4);
  float*  hb_p4  = (float*)alloc((size_t)NEVT * 32 * 4);
  float*  hb_ch  = (float*)alloc((size_t)NEVT * 32 * 4);
  float*  hb_dm  = (float*)alloc((size_t)NEVT * 32 * 4);
  half_t* warena = (half_t*)alloc((size_t)8 << 20);
  size_t  woff   = 0;

  // chunked edge buffers sized to remaining workspace (deterministic in ws_size)
  size_t used = cur - base;
  int CE = 64;
  while (CE > 16) {
    size_t need = (size_t)CE * NPART * 19 * 2 * (512 + 3 * 256) + 4096;
    if (used + need + 4096 <= ws_size) break;
    CE >>= 1;
  }
  size_t  hrows = (size_t)CE * NPART * 19;
  half_t* h0 = (half_t*)alloc(hrows * 512 * 2);
  half_t* h1 = (half_t*)alloc(hrows * 256 * 2);
  half_t* h2 = (half_t*)alloc(hrows * 256 * 2);
  half_t* h3 = (half_t*)alloc(hrows * 256 * 2);

  auto walloc = [&](size_t elems) -> half_t* {
    half_t* p = warena + woff;
    woff += (elems + 127) & ~(size_t)127;
    return p;
  };
  auto wconv = [&](const float* W, int K, int Nv, int Kp, int Np) -> half_t* {
    half_t* p = walloc((size_t)Kp * Np);
    int tot = Kp * Np;
    convert_w_kernel<<<(tot + 255) / 256, 256, 0, stream>>>(W, p, K, Nv, Kp, Np, tot);
    return p;
  };
  // dispatch: 64-wide tiles when possible, else 32-wide
  auto gemm = [&](const half_t* A, int lda, const half_t* Wt, int Kp, int M, int Np,
                  float* C32, int ldc32, half_t* C16, int ldc16,
                  int epi, const float* E0, const float* E1, const float* E2, const float* E3,
                  int Nvalid) {
    if (Np % 64 == 0) {
      dim3 grid(M / 32, Np / 64);
      wmma_gemm_kernel<4><<<grid, dim3(32), 0, stream>>>(A, lda, Wt, Kp, Kp, Np,
                                                         C32, ldc32, C16, ldc16,
                                                         epi, E0, E1, E2, E3, Nvalid);
    } else {
      dim3 grid(M / 32, Np / 32);
      wmma_gemm_kernel<2><<<grid, dim3(32), 0, stream>>>(A, lda, Wt, Kp, Kp, Np,
                                                         C32, ldc32, C16, ldc16,
                                                         epi, E0, E1, E2, E3, Nvalid);
    }
  };

  // ---- input BN: fts_a[20480,14] ----
  { int tot = NODES * NFEAT;
    bn_input_kernel<<<(tot + 255) / 256, 256, 0, stream>>>(
        gnnfeats, P(2), P(3), P(4), P(5), fts_a, tot); }

  // ---- EdgeConv stack ----
  const int c0s[5] = {64, 128, 256, 256, 256};
  const int c1s[5] = {64, 128, 256, 256, 128};
  const int c2s[5] = {64, 256, 256, 256, 20};
  const int kEs[5] = {2, 4, 8, 16, 19};  // k=min(k, N-1)
  int inC = NFEAT;
  float* ftsCur = fts_a; float* ftsNext = fts_b;

  for (int ci = 0; ci < 5; ++ci) {
    int pb = 28 + 20 * ci;
    int c0 = c0s[ci], c1 = c1s[ci], c2 = c2s[ci], kE = kEs[ci];
    int Kp0 = pad32i(2 * inC);
    int Np2 = pad32i(c2);   // 32-aligned output cols (c2=20 -> 32)

    const float* pts = (ci == 0) ? gnnpos : ftsCur;
    int D = (ci == 0) ? 2 : inC;
    knn_kernel<<<NEVT, 32, 0, stream>>>(pts, D, kE, nbr);

    half_t* Wt0 = wconv(P(pb + 0),  2 * inC, c0, Kp0, c0);
    half_t* Wt1 = wconv(P(pb + 5),  c0, c1, c0, c1);
    half_t* Wt2 = wconv(P(pb + 10), c1, c2, c1, Np2);
    half_t* Wts = wconv(P(pb + 19), inC, c2, pad32i(inC), Np2);

    // skip = BN(fts @ skip_w), full batch
    { int tot = NODES * pad32i(inC);
      pack_half_kernel<<<(tot + 255) / 256, 256, 0, stream>>>(ftsCur, inC, act16, pad32i(inC), tot); }
    gemm(act16, pad32i(inC), Wts, pad32i(inC), NODES, Np2,
         skip, Np2, nullptr, 0, 1, P(pb + 15), P(pb + 16), P(pb + 17), P(pb + 18), c2);

    for (int e0 = 0; e0 < NEVT; e0 += CE) {
      int M = CE * NPART * kE;
      const float* gf = ftsCur + (size_t)e0 * NPART * inC;
      const int*   gn = nbr   + (size_t)e0 * NPART * kE;
      { int tot = M * Kp0;
        edge_build_kernel<<<(tot + 255) / 256, 256, 0, stream>>>(gf, gn, inC, kE, h0, Kp0, tot); }
      gemm(h0, Kp0, Wt0, Kp0, M, c0, nullptr, 0, h1, c0,
           0, P(pb + 1), P(pb + 2), P(pb + 3), P(pb + 4), c0);
      gemm(h1, c0, Wt1, c0, M, c1, nullptr, 0, h2, c1,
           0, P(pb + 6), P(pb + 7), P(pb + 8), P(pb + 9), c1);
      gemm(h2, c1, Wt2, c1, M, Np2, nullptr, 0, h3, Np2,
           0, P(pb + 11), P(pb + 12), P(pb + 13), P(pb + 14), c2);
      int tot = CE * NPART * c2;
      aggr_kernel<<<(tot + 255) / 256, 256, 0, stream>>>(h3, Np2, kE, c2, skip, Np2,
                                                         ftsNext, e0 * NPART, tot);
    }
    float* tmp = ftsCur; ftsCur = ftsNext; ftsNext = tmp;
    inC = c2;
  }

  // ---- encoder input [B,680] ----
  { int tot = NEVT * 680;
    build_nxt_kernel<<<(tot + 255) / 256, 256, 0, stream>>>(gnnfeats, ftsCur, nxt, tot); }

  // ---- FFN driver (LN -> GEMM(+bias,ELU) x4; elu(LN); LN; GEMM(+bias)) ----
  auto run_ffn = [&](const float* xin, int Lin, int pb, float* yout, int NpF, int NvF) {
    const float* x = xin; int L = Lin;
    float* bufs[2] = {x0, x1};
    for (int l = 0; l < 4; ++l) {
      int Kp = pad32i(L);
      layernorm_kernel<<<NEVT, 32, 0, stream>>>(x, L, L, P(pb + 10 + 2 * l), P(pb + 11 + 2 * l),
                                                t16, Kp, 1, 0);
      half_t* Wt = wconv(P(pb + 2 * l), L, 128, Kp, 128);
      float* y = bufs[l & 1];
      gemm(t16, Kp, Wt, Kp, NEVT, 128, y, 128, nullptr, 0,
           2, P(pb + 1 + 2 * l), nullptr, nullptr, nullptr, 128);
      x = y; L = 128;
    }
    layernorm_kernel<<<NEVT, 32, 0, stream>>>(x, 128, 128, P(pb + 18), P(pb + 19), x2, 128, 0, 1);
    layernorm_kernel<<<NEVT, 32, 0, stream>>>(x2, 128, 128, P(pb + 20), P(pb + 21), t16, 128, 1, 0);
    half_t* Wt4 = wconv(P(pb + 8), 128, NvF, 128, NpF);
    gemm(t16, 128, Wt4, 128, NEVT, NpF, yout, NpF, nullptr, 0,
         3, P(pb + 9), nullptr, nullptr, nullptr, NvF);
  };

  run_ffn(nxt,   680, 150, enc_o, 128, 128);  // enc
  run_ffn(enc_o, 128, 172, hb_is, 32, 1);     // istau
  run_ffn(enc_o, 128, 194, hb_p4, 32, 4);     // p4
  run_ffn(enc_o, 128, 6,   hb_ch, 32, 2);     // charge
  run_ffn(enc_o, 128, 128, hb_dm, 32, 6);     // dmode

  finalize_kernel<<<NEVT, 32, 0, stream>>>(ftsCur, hb_is, hb_p4, hb_ch, hb_dm, (float*)d_out);
}
